// FrequencyOnlyBranchFast_75814762709383
// MI455X (gfx1250) — matrix operand order repaired and checked by reading, not compiled, not run on hardware
//
#include <hip/hip_runtime.h>
#include <math.h>

typedef __bf16 v16bf __attribute__((ext_vector_type(16)));
typedef float  v8f   __attribute__((ext_vector_type(8)));

union FragBF { v16bf v; unsigned u[8]; };

#define PI2F 6.28318530717958647692f

// ---------------------------------------------------------------------------
// Kernel 0: zero wsum[1024] + stats[2] (contiguous)
// ---------------------------------------------------------------------------
__global__ void zero_acc_kernel(float* wsum) {
    int i = blockIdx.x * blockDim.x + threadIdx.x;
    if (i < 1026) wsum[i] = 0.0f;
}

// ---------------------------------------------------------------------------
// Kernel 1: grayscale -> patchify -> 2D 32x32 DFT magnitude -> radial gate
// one block (256 thr) per patch; LDS twiddle table; outputs fp32 + bf16
// ---------------------------------------------------------------------------
__global__ void __launch_bounds__(256)
fft_mag_kernel(const float* __restrict__ x,
               const float* __restrict__ lfg, const float* __restrict__ mfg,
               const float* __restrict__ hfg,
               float* __restrict__ magf, __bf16* __restrict__ magbf) {
    __shared__ float xs[1024];
    __shared__ float f1re[1024];
    __shared__ float f1im[1024];
    __shared__ float ct[32], st[32];

    const int t  = threadIdx.x;
    const int pb = blockIdx.x;          // 0..1023 patch id
    const int b  = pb >> 6;             // image
    const int l  = pb & 63;             // patch within image
    const int gy = l >> 3, gx = l & 7;

    if (t < 32) {
        float ang = -PI2F * (float)t * (1.0f / 32.0f);
        ct[t] = __cosf(ang);
        st[t] = __sinf(ang);
    }

    const float* xb = x + (size_t)b * 3 * 65536;
    #pragma unroll
    for (int i = 0; i < 4; ++i) {
        int idx = t + i * 256;
        int py = idx >> 5, px = idx & 31;
        int off = (gy * 32 + py) * 256 + (gx * 32 + px);
        xs[idx] = 0.299f * xb[off] + 0.587f * xb[65536 + off] + 0.114f * xb[131072 + off];
    }
    __syncthreads();

    // row DFT (along x)
    #pragma unroll
    for (int i = 0; i < 4; ++i) {
        int idx = t + i * 256;
        int row = idx >> 5, k = idx & 31;
        float re = 0.f, im = 0.f;
        #pragma unroll 8
        for (int n = 0; n < 32; ++n) {
            float v = xs[row * 32 + n];
            int j = (k * n) & 31;
            re += v * ct[j];
            im += v * st[j];
        }
        f1re[idx] = re;
        f1im[idx] = im;
    }
    __syncthreads();

    const float sl = 1.f / (1.f + __expf(-lfg[0]));
    const float sm = 1.f / (1.f + __expf(-mfg[0]));
    const float sh = 1.f / (1.f + __expf(-hfg[0]));
    const float rm = 21.92031022f;       // 15.5*sqrt(2)

    // column DFT (along y), magnitude, ortho norm, gate
    #pragma unroll
    for (int i = 0; i < 4; ++i) {
        int idx = t + i * 256;
        int ky = idx >> 5, kx = idx & 31;
        float re = 0.f, im = 0.f;
        #pragma unroll 8
        for (int m = 0; m < 32; ++m) {
            float ar = f1re[m * 32 + kx];
            float ai = f1im[m * 32 + kx];
            int j = (ky * m) & 31;
            float c = ct[j], s = st[j];
            re += ar * c - ai * s;
            im += ar * s + ai * c;
        }
        float mag = sqrtf(re * re + im * im) * (1.0f / 32.0f);
        float dy = (float)ky - 15.5f, dx = (float)kx - 15.5f;
        float r = sqrtf(dy * dy + dx * dx);
        float g = (r <= rm * (1.0f / 3.0f)) ? sl
                : ((r <= rm * (2.0f / 3.0f)) ? sm : sh);
        mag *= g;
        size_t o = (size_t)pb * 1024 + idx;
        magf[o]  = mag;
        magbf[o] = (__bf16)mag;
    }
}

// ---------------------------------------------------------------------------
// Kernel 2: fp32 weights -> bf16, transposed to [N][K] (contiguous-K fragments)
// ---------------------------------------------------------------------------
__global__ void convert_w_kernel(const float* __restrict__ W1, const float* __restrict__ W2,
                                 __bf16* __restrict__ W1T, __bf16* __restrict__ W2T) {
    int stride = gridDim.x * blockDim.x;
    for (int j = blockIdx.x * blockDim.x + threadIdx.x; j < 1024 * 1536; j += stride) {
        int n = j >> 10;                 // W1T: [1536][1024]
        int k = j & 1023;
        W1T[j] = (__bf16)W1[k * 1536 + n];
        int n2 = j / 1536;               // W2T: [1024][1536]
        int k2 = j - n2 * 1536;
        W2T[j] = (__bf16)W2[k2 * 1024 + n2];
    }
}

// ---------------------------------------------------------------------------
// WMMA fragment loaders (16-bit A 16x32 / B 32x16, contiguous k = 16*half + s in both, wave32)
// ---------------------------------------------------------------------------
__device__ __forceinline__ v16bf load_a_frag(const __bf16* A, int row, int K, int k0, int lane) {
    const int half = lane >> 4;
    const unsigned* p = (const unsigned*)(A + (size_t)row * K + k0);
    FragBF f;
    #pragma unroll
    for (int q = 0; q < 8; ++q) {
        int kk = 2 * q + half * 16;
        f.u[q] = p[kk >> 1];
    }
    return f.v;
}

__device__ __forceinline__ v16bf load_b_frag(const __bf16* Bt, int col, int K, int k0, int lane) {
    const int half = lane >> 4;
    const unsigned* p = (const unsigned*)(Bt + (size_t)col * K + k0 + half * 16);
    FragBF f;
    #pragma unroll
    for (int q = 0; q < 8; ++q) f.u[q] = p[q];
    return f.v;
}

// ---------------------------------------------------------------------------
// Kernel 3: GEMM1  H = relu(mag @ W1 + b1)   [1024,1024]x[1024,1536] -> bf16
// one wave per 16x64 strip (A frag reused across 4 WMMAs); 8 waves/block
// ---------------------------------------------------------------------------
__global__ void __launch_bounds__(256)
gemm1_kernel(const __bf16* __restrict__ A, const __bf16* __restrict__ Bt,
             const float* __restrict__ b1, __bf16* __restrict__ H) {
    const int N = 1536, K = 1024;
    const int lane = threadIdx.x & 31;
    const int wave = threadIdx.x >> 5;
    const int gid  = blockIdx.x * 8 + wave;       // 64 * 24 = 1536 strips
    const int tm = gid / 24, tg = gid % 24;
    const int rowbase = tm * 16, colbase = tg * 64;

    v8f acc0 = {}, acc1 = {}, acc2 = {}, acc3 = {};
    const int arow = rowbase + (lane & 15);
    const int bcol = colbase + (lane & 15);
    for (int k0 = 0; k0 < K; k0 += 32) {
        v16bf a  = load_a_frag(A, arow, K, k0, lane);
        v16bf b0 = load_b_frag(Bt, bcol,      K, k0, lane);
        v16bf b1f = load_b_frag(Bt, bcol + 16, K, k0, lane);
        v16bf b2f = load_b_frag(Bt, bcol + 32, K, k0, lane);
        v16bf b3f = load_b_frag(Bt, bcol + 48, K, k0, lane);
        acc0 = __builtin_amdgcn_wmma_f32_16x16x32_bf16(false, a, false, b0,  (short)0, acc0, false, false);
        acc1 = __builtin_amdgcn_wmma_f32_16x16x32_bf16(false, a, false, b1f, (short)0, acc1, false, false);
        acc2 = __builtin_amdgcn_wmma_f32_16x16x32_bf16(false, a, false, b2f, (short)0, acc2, false, false);
        acc3 = __builtin_amdgcn_wmma_f32_16x16x32_bf16(false, a, false, b3f, (short)0, acc3, false, false);
    }
    const int half = lane >> 4;
    v8f accs[4] = {acc0, acc1, acc2, acc3};
    #pragma unroll
    for (int j = 0; j < 4; ++j) {
        int n = colbase + 16 * j + (lane & 15);
        float bias = b1[n];
        #pragma unroll
        for (int v = 0; v < 8; ++v) {
            int r = rowbase + v + 8 * half;
            float val = accs[j][v] + bias;
            H[(size_t)r * N + n] = (__bf16)fmaxf(val, 0.0f);
        }
    }
}

// ---------------------------------------------------------------------------
// Kernel 4: GEMM2 + fused sigmoid * mag, cross-lane row-sum -> wsum[1024]
// one wave per 16x64 strip; 8 waves/block
// ---------------------------------------------------------------------------
__global__ void __launch_bounds__(256)
gemm2_kernel(const __bf16* __restrict__ A, const __bf16* __restrict__ Bt,
             const float* __restrict__ b2, const float* __restrict__ magf,
             float* __restrict__ wsum) {
    const int K = 1536;
    const int lane = threadIdx.x & 31;
    const int wave = threadIdx.x >> 5;
    const int gid  = blockIdx.x * 8 + wave;       // 64 * 16 = 1024 strips
    const int tm = gid >> 4, tg = gid & 15;
    const int rowbase = tm * 16, colbase = tg * 64;

    v8f acc0 = {}, acc1 = {}, acc2 = {}, acc3 = {};
    const int arow = rowbase + (lane & 15);
    const int bcol = colbase + (lane & 15);
    for (int k0 = 0; k0 < K; k0 += 32) {
        v16bf a  = load_a_frag(A, arow, K, k0, lane);
        v16bf b0 = load_b_frag(Bt, bcol,      K, k0, lane);
        v16bf b1f = load_b_frag(Bt, bcol + 16, K, k0, lane);
        v16bf b2f = load_b_frag(Bt, bcol + 32, K, k0, lane);
        v16bf b3f = load_b_frag(Bt, bcol + 48, K, k0, lane);
        acc0 = __builtin_amdgcn_wmma_f32_16x16x32_bf16(false, a, false, b0,  (short)0, acc0, false, false);
        acc1 = __builtin_amdgcn_wmma_f32_16x16x32_bf16(false, a, false, b1f, (short)0, acc1, false, false);
        acc2 = __builtin_amdgcn_wmma_f32_16x16x32_bf16(false, a, false, b2f, (short)0, acc2, false, false);
        acc3 = __builtin_amdgcn_wmma_f32_16x16x32_bf16(false, a, false, b3f, (short)0, acc3, false, false);
    }
    const int half = lane >> 4;
    v8f accs[4] = {acc0, acc1, acc2, acc3};
    #pragma unroll
    for (int j = 0; j < 4; ++j) {
        int n = colbase + 16 * j + (lane & 15);
        float bias = b2[n];
        #pragma unroll
        for (int v = 0; v < 8; ++v) {
            int r = rowbase + v + 8 * half;
            float val = accs[j][v] + bias;
            float s = 1.0f / (1.0f + __expf(-val));
            float mod = s * magf[(size_t)r * 1024 + n];
            // sum over the 16 lanes holding row r (stays inside each 16-lane half)
            mod += __shfl_xor(mod, 1, 32);
            mod += __shfl_xor(mod, 2, 32);
            mod += __shfl_xor(mod, 4, 32);
            mod += __shfl_xor(mod, 8, 32);
            if ((lane & 15) == 0) atomicAdd(&wsum[r], mod);
        }
    }
}

// ---------------------------------------------------------------------------
// Kernel 5: x_rec = gray * (1 + w[patch]); scalar sum / sumsq reduction
// ---------------------------------------------------------------------------
__global__ void __launch_bounds__(256)
stats_kernel(const float* __restrict__ x, const float* __restrict__ wsum,
             float* __restrict__ xrec, float* __restrict__ stats) {
    __shared__ float ssum[256];
    __shared__ float ssq[256];
    const int idx = blockIdx.x * 256 + threadIdx.x;   // 1,048,576 pixels
    const int b  = idx >> 16;
    const int hw = idx & 65535;
    const int y  = hw >> 8, xc = hw & 255;
    const int patch = b * 64 + (y >> 5) * 8 + (xc >> 5);
    const float w = wsum[patch] * (1.0f / 1024.0f);
    const float* xb = x + (size_t)b * 3 * 65536;
    float xg = 0.299f * xb[hw] + 0.587f * xb[65536 + hw] + 0.114f * xb[131072 + hw];
    float xr = xg * (1.0f + w);
    xrec[idx] = xr;

    ssum[threadIdx.x] = xr;
    ssq[threadIdx.x]  = xr * xr;
    __syncthreads();
    for (int s = 128; s > 0; s >>= 1) {
        if (threadIdx.x < s) {
            ssum[threadIdx.x] += ssum[threadIdx.x + s];
            ssq[threadIdx.x]  += ssq[threadIdx.x + s];
        }
        __syncthreads();
    }
    if (threadIdx.x == 0) {
        atomicAdd(&stats[0], ssum[0]);
        atomicAdd(&stats[1], ssq[0]);
    }
}

// ---------------------------------------------------------------------------
// Kernel 6: per-channel BN scale (BN factors through 1x1 conv), mu
// ---------------------------------------------------------------------------
__global__ void params_kernel(const float* __restrict__ stats,
                              const float* __restrict__ w_proj,
                              const float* __restrict__ gamma,
                              float* __restrict__ params) {
    const float invN = 1.0f / 1048576.0f;
    float mu  = stats[0] * invN;
    float var = stats[1] * invN - mu * mu;
    int c = threadIdx.x;
    if (c < 64) {
        float wp = w_proj[c];
        params[c] = wp * gamma[c] * rsqrtf(wp * wp * var + 1e-5f);
    }
    if (c == 0) params[64] = mu;
}

// ---------------------------------------------------------------------------
// Kernel 7: streaming 256MB output: relu(scale_c*(xrec-mu)+beta_c), NT stores
// ---------------------------------------------------------------------------
__global__ void __launch_bounds__(256)
final_kernel(const float* __restrict__ xrec, const float* __restrict__ params,
             const float* __restrict__ beta, float* __restrict__ out) {
    __shared__ float sc[64];
    __shared__ float sb[64];
    if (threadIdx.x < 64) {
        sc[threadIdx.x] = params[threadIdx.x];
        sb[threadIdx.x] = beta[threadIdx.x];
    }
    __syncthreads();
    const float mu = params[64];
    const int idx = blockIdx.x * 256 + threadIdx.x;
    const int b  = idx >> 16;
    const int hw = idx & 65535;
    const float d = xrec[idx] - mu;
    float* ob = out + (size_t)b * 64 * 65536 + hw;
    #pragma unroll
    for (int c = 0; c < 64; ++c) {
        float v = fmaxf(sc[c] * d + sb[c], 0.0f);
        __builtin_nontemporal_store(v, ob + (size_t)c * 65536);
    }
}

// ---------------------------------------------------------------------------
extern "C" void kernel_launch(void* const* d_in, const int* in_sizes, int n_in,
                              void* d_out, int out_size, void* d_ws, size_t ws_size,
                              hipStream_t stream) {
    const float* x      = (const float*)d_in[0];
    const float* W1     = (const float*)d_in[1];
    const float* b1     = (const float*)d_in[2];
    const float* W2     = (const float*)d_in[3];
    const float* b2     = (const float*)d_in[4];
    const float* w_proj = (const float*)d_in[5];
    const float* gamma  = (const float*)d_in[6];
    const float* beta   = (const float*)d_in[7];
    const float* lfg    = (const float*)d_in[8];
    const float* mfg    = (const float*)d_in[9];
    const float* hfg    = (const float*)d_in[10];
    float* out = (float*)d_out;

    char* ws = (char*)d_ws;
    float*  magf  = (float*)(ws);                          // 4 MiB
    __bf16* magbf = (__bf16*)(ws + (4u << 20));            // 2 MiB
    __bf16* W1T   = (__bf16*)(ws + (6u << 20));            // 3 MiB [1536][1024]
    __bf16* W2T   = (__bf16*)(ws + (9u << 20));            // 3 MiB [1024][1536]
    __bf16* Hbf   = (__bf16*)(ws + (12u << 20));           // 3 MiB [1024][1536]
    float*  wsum  = (float*)(ws + (15u << 20));            // 1024 floats
    float*  stats = wsum + 1024;                           // 2 floats (contig)
    float*  params = (float*)(ws + (15u << 20) + 8192);    // 65 floats
    float*  xrec  = (float*)(ws + (16u << 20));            // 4 MiB

    zero_acc_kernel<<<5, 256, 0, stream>>>(wsum);
    fft_mag_kernel<<<1024, 256, 0, stream>>>(x, lfg, mfg, hfg, magf, magbf);
    convert_w_kernel<<<1536, 256, 0, stream>>>(W1, W2, W1T, W2T);
    gemm1_kernel<<<192, 256, 0, stream>>>(magbf, W1T, b1, Hbf);      // 64*24 strips
    gemm2_kernel<<<128, 256, 0, stream>>>(Hbf, W2T, b2, magf, wsum); // 64*16 strips
    stats_kernel<<<4096, 256, 0, stream>>>(x, wsum, xrec, stats);
    params_kernel<<<1, 64, 0, stream>>>(stats, w_proj, gamma, params);
    final_kernel<<<4096, 256, 0, stream>>>(xrec, params, beta, out);
}
